// PatchAttention_24902220382269
// MI455X (gfx1250) — matrix operand order repaired and checked by reading, not compiled, not run on hardware
//
#include <hip/hip_runtime.h>

typedef __attribute__((ext_vector_type(16))) __bf16 v16bf;
typedef __attribute__((ext_vector_type(2)))  __bf16 v2bf;
typedef __attribute__((ext_vector_type(8)))  float  v8f;
typedef __attribute__((ext_vector_type(4)))  unsigned int u32x4;
typedef __attribute__((ext_vector_type(4)))  float  f32x4;

#define DI __device__ __forceinline__

union Frag {
  v16bf v;
  u32x4 q[2];
  unsigned int w[8];
};

constexpr int C  = 128;
constexpr int S  = 16;
constexpr int H  = 512;
constexpr int W  = 512;
constexpr int L  = 256;      // tokens per patch
constexpr int GW = 32;       // patches per row
constexpr int NPB = 1024;    // patches per batch image
constexpr int HW = H * W;

constexpr int PPITCH = 136;  // row pitch (bf16 elems) for [l][c]/[l][d]/[m][d] buffers
constexpr int VPITCH = 264;  // row pitch for V[d][m]

constexpr int OFF_P   = 0;
constexpr int OFF_Q   = OFF_P + L * PPITCH;   // reused as Y in phase 3
constexpr int OFF_K   = OFF_Q + L * PPITCH;
constexpr int OFF_V   = OFF_K + L * PPITCH;
constexpr int ATT_PER_WAVE = 16 * 32;
constexpr int OFF_ATT = OFF_V + C * VPITCH;
constexpr int LDS_ELEMS = OFF_ATT + 16 * ATT_PER_WAVE;
static_assert(LDS_ELEMS * 2 <= 320 * 1024, "LDS over 320KB WGP budget");

// f32 -> bf16 packing. Prefer the builtin if declared; otherwise try the
// native gfx1250 instruction via inline asm (1 VALU op per pair instead of ~5).
#if __has_builtin(__builtin_amdgcn_cvt_pk_bf16_f32)
DI unsigned int pack2(float a, float b) {
  union { v2bf v; unsigned int u; } cv;
  cv.v = __builtin_amdgcn_cvt_pk_bf16_f32(a, b);
  return cv.u;
}
#else
DI unsigned int pack2(float a, float b) {
  unsigned int r;
  asm("v_cvt_pk_bf16_f32 %0, %1, %2" : "=v"(r) : "v"(a), "v"(b));
  return r;
}
#endif
DI unsigned short f2bf(float f) {
  return (unsigned short)pack2(f, f);   // low half = bf16(f)
}

DI v8f wmma_bf16(const Frag& a, const Frag& b, v8f c) {
  return __builtin_amdgcn_wmma_f32_16x16x32_bf16(false, a.v, false, b.v,
                                                 (short)0, c, false, false);
}

// Load a 16x32 A-fragment of a row-major f32 weight matrix (rows = output dim,
// K = input dim), converting to bf16 in registers.
DI void load_wfrag(const float* __restrict__ Wp, int row, int kf, int half, Frag& a) {
  const float* wr = Wp + row * C + kf * 32 + half * 16;
  f32x4 f0 = *(const f32x4*)(wr);
  f32x4 f1 = *(const f32x4*)(wr + 4);
  f32x4 f2 = *(const f32x4*)(wr + 8);
  f32x4 f3 = *(const f32x4*)(wr + 12);
  a.w[0] = pack2(f0.x, f0.y); a.w[1] = pack2(f0.z, f0.w);
  a.w[2] = pack2(f1.x, f1.y); a.w[3] = pack2(f1.z, f1.w);
  a.w[4] = pack2(f2.x, f2.y); a.w[5] = pack2(f2.z, f2.w);
  a.w[6] = pack2(f3.x, f3.y); a.w[7] = pack2(f3.z, f3.w);
}

DI void load_bias8(const float* __restrict__ bp, int base, int half, float bb[8]) {
  const float* bpp = bp + base + half * 8;
  f32x4 b0 = *(const f32x4*)bpp;
  f32x4 b1 = *(const f32x4*)(bpp + 4);
  bb[0] = b0.x; bb[1] = b0.y; bb[2] = b0.z; bb[3] = b0.w;
  bb[4] = b1.x; bb[5] = b1.y; bb[6] = b1.z; bb[7] = b1.w;
}

__global__ __launch_bounds__(512, 1)
void patch_attn_kernel(const float* __restrict__ x,
                       const float* __restrict__ Wq, const float* __restrict__ bq,
                       const float* __restrict__ Wk, const float* __restrict__ bk,
                       const float* __restrict__ Wv, const float* __restrict__ bv,
                       const float* __restrict__ Wo, const float* __restrict__ bo,
                       float* __restrict__ out)
{
  __shared__ unsigned short lds[LDS_ELEMS];

  const int n   = blockIdx.x;
  const int b   = n / NPB;
  const int pid = n % NPB;
  const int ph  = pid / GW, pw = pid % GW;
  const float* xb = x   + (size_t)b * C * HW;
  float*       ob = out + (size_t)b * C * HW;
  const int pbase = (ph * S) * W + pw * S;     // patch origin inside a channel plane

  const int tid  = threadIdx.x;
  const int lane = tid & 31;
  const int wave = tid >> 5;                   // 16 waves
  const int m16  = lane & 15;
  const int half = lane >> 4;
  const int l0   = wave * 16;                  // this wave's l-tile (phase 2)

  // ---------------- Phase 0: x patch -> bf16 Pt[l][c] ----------------
  for (int e = tid; e < C * L; e += 512) {
    int c = e >> 8, l = e & 255;
    float f = xb[c * HW + pbase + (l >> 4) * W + (l & 15)];
    lds[OFF_P + l * PPITCH + c] = f2bf(f);
  }
  __syncthreads();

  // ---------------- Phase 1: Q/K/V projections ----------------
  // Unit 0 (all 16 waves): d-tile (wave&7) of Q (waves 0-7) or K (waves 8-15).
  // Unit 1 (waves 0-7):    d-tile (wave&7) of V.
  // Each unit converts its weight slab ONCE, then sweeps all 16 l-tiles.
  {
    const int   dt0   = wave & 7;
    const bool  isQ   = (wave < 8);
    const float* W0   = isQ ? Wq : Wk;
    const float* b0p  = isQ ? bq : bk;
    const int   qkOff = isQ ? OFF_Q : OFF_K;

    Frag wa0[4];
#pragma unroll
    for (int kf = 0; kf < 4; ++kf)
      load_wfrag(W0, dt0 * 16 + m16, kf, half, wa0[kf]);
    float wb0[8];
    load_bias8(b0p, dt0 * 16, half, wb0);

    const bool hasV = (wave < 8);
    Frag wav[4];
    float wbv[8];
    if (hasV) {
#pragma unroll
      for (int kf = 0; kf < 4; ++kf)
        load_wfrag(Wv, dt0 * 16 + m16, kf, half, wav[kf]);
      load_bias8(bv, dt0 * 16, half, wbv);
    }

    for (int lt = 0; lt < 16; ++lt) {
      Frag pb[4];
#pragma unroll
      for (int kf = 0; kf < 4; ++kf) {
        int base = OFF_P + (lt * 16 + m16) * PPITCH + kf * 32 + half * 16;
        pb[kf].q[0] = *(const u32x4*)&lds[base];
        pb[kf].q[1] = *(const u32x4*)&lds[base + 8];
      }

      // Unit 0: Q or K, transposed store -> [l][d] (8 consecutive d per lane).
      {
        v8f acc = {};
#pragma unroll
        for (int kf = 0; kf < 4; ++kf) acc = wmma_bf16(wa0[kf], pb[kf], acc);
        int dst = qkOff + (lt * 16 + m16) * PPITCH + dt0 * 16 + half * 8;
        u32x4 st;
        st.x = pack2(acc[0] + wb0[0], acc[1] + wb0[1]);
        st.y = pack2(acc[2] + wb0[2], acc[3] + wb0[3]);
        st.z = pack2(acc[4] + wb0[4], acc[5] + wb0[5]);
        st.w = pack2(acc[6] + wb0[6], acc[7] + wb0[7]);
        *(u32x4*)&lds[dst] = st;
      }

      // Unit 1: V, natural store -> V[d][m].
      if (hasV) {
        v8f av = {};
#pragma unroll
        for (int kf = 0; kf < 4; ++kf) av = wmma_bf16(wav[kf], pb[kf], av);
#pragma unroll
        for (int r = 0; r < 8; ++r) {
          int dp = dt0 * 16 + half * 8 + r;
          lds[OFF_V + dp * VPITCH + lt * 16 + m16] = f2bf(av[r] + wbv[r]);
        }
      }
    }
  }
  __syncthreads();

  // ---------------- Phase 2: attention (single pass, deferred normalization) ----------------
  Frag qa[4];
#pragma unroll
  for (int kf = 0; kf < 4; ++kf) {
    int base = OFF_Q + (l0 + m16) * PPITCH + kf * 32 + half * 16;
    qa[kf].q[0] = *(const u32x4*)&lds[base];
    qa[kf].q[1] = *(const u32x4*)&lds[base + 8];
  }

  v8f o[8];
#pragma unroll
  for (int dt = 0; dt < 8; ++dt) o[dt] = (v8f){};
  float rs[8] = {0.f, 0.f, 0.f, 0.f, 0.f, 0.f, 0.f, 0.f};
  const int attW = OFF_ATT + wave * ATT_PER_WAVE;
  constexpr float SM_SCALE = 1.4426950408889634f / 11.313708498984761f; // log2(e)/sqrt(C)

  for (int mc = 0; mc < 8; ++mc) {
#pragma unroll
    for (int t = 0; t < 2; ++t) {
      int mt = mc * 2 + t;
      Frag kb[4];
#pragma unroll
      for (int kf = 0; kf < 4; ++kf) {
        int base = OFF_K + (mt * 16 + m16) * PPITCH + kf * 32 + half * 16;
        kb[kf].q[0] = *(const u32x4*)&lds[base];
        kb[kf].q[1] = *(const u32x4*)&lds[base + 8];
      }
      v8f s = {};
#pragma unroll
      for (int kf = 0; kf < 4; ++kf) s = wmma_bf16(qa[kf], kb[kf], s);
#pragma unroll
      for (int r = 0; r < 8; ++r) {
        float e = __builtin_exp2f(s[r] * SM_SCALE);
        rs[r] += e;
        lds[attW + (half * 8 + r) * 32 + t * 16 + m16] = f2bf(e);
      }
    }
    // same-wave RAW on per-wave LDS scratch (LDS is in-order, wait for stores)
    asm volatile("s_wait_dscnt 0" ::: "memory");
    Frag aw;
    {
      int base = attW + m16 * 32 + half * 16;
      aw.q[0] = *(const u32x4*)&lds[base];
      aw.q[1] = *(const u32x4*)&lds[base + 8];
    }
#pragma unroll
    for (int dt = 0; dt < 8; ++dt) {
      Frag vb;
      int base = OFF_V + (dt * 16 + m16) * VPITCH + mc * 32 + half * 16;
      vb.q[0] = *(const u32x4*)&lds[base];
      vb.q[1] = *(const u32x4*)&lds[base + 8];
      o[dt] = wmma_bf16(aw, vb, o[dt]);
    }
  }

  // Row-sum reduction across the 16 lanes sharing a row (wave32 halves stay disjoint).
#pragma unroll
  for (int r = 0; r < 8; ++r) {
    float v = rs[r];
    v += __shfl_xor(v, 1, 32);
    v += __shfl_xor(v, 2, 32);
    v += __shfl_xor(v, 4, 32);
    v += __shfl_xor(v, 8, 32);
    rs[r] = 1.0f / v;
  }

  // Normalize O and store as Y[l][d] bf16, reusing Q's LDS space (own rows only).
#pragma unroll
  for (int dt = 0; dt < 8; ++dt) {
#pragma unroll
    for (int r = 0; r < 8; ++r) {
      int lp = l0 + half * 8 + r;
      lds[OFF_Q + lp * PPITCH + dt * 16 + m16] = f2bf(o[dt][r] * rs[r]);
    }
  }
  __syncthreads();

  // ---------------- Phase 3: output projection + bias + residual ----------------
  // 8 c-tile units x 2 waves each; each wave converts its Wo slab once and
  // sweeps 8 l-tiles of Y.
  {
    const int ct = wave >> 1;
    const int ltbase = (wave & 1) * 8;

    Frag wo[4];
#pragma unroll
    for (int kf = 0; kf < 4; ++kf)
      load_wfrag(Wo, ct * 16 + m16, kf, half, wo[kf]);
    float bb[8];
    load_bias8(bo, ct * 16, half, bb);

    for (int li = 0; li < 8; ++li) {
      int lt = ltbase + li;
      Frag yb[4];
#pragma unroll
      for (int kf = 0; kf < 4; ++kf) {
        int base = OFF_Q + (lt * 16 + m16) * PPITCH + kf * 32 + half * 16;
        yb[kf].q[0] = *(const u32x4*)&lds[base];
        yb[kf].q[1] = *(const u32x4*)&lds[base + 8];
      }
      v8f acc = {};
#pragma unroll
      for (int kf = 0; kf < 4; ++kf) acc = wmma_bf16(wo[kf], yb[kf], acc);

      int lpp = lt * 16 + m16;
      int gcol = pbase + (lpp >> 4) * W + (lpp & 15);
#pragma unroll
      for (int r = 0; r < 8; ++r) {
        int cp = ct * 16 + half * 8 + r;
        size_t off = (size_t)cp * HW + gcol;
        ob[off] = xb[off] + acc[r] + bb[r];   // identity residual
      }
    }
  }
}

extern "C" void kernel_launch(void* const* d_in, const int* in_sizes, int n_in,
                              void* d_out, int out_size, void* d_ws, size_t ws_size,
                              hipStream_t stream) {
  const float* x  = (const float*)d_in[0];
  const float* Wq = (const float*)d_in[1];
  const float* bq = (const float*)d_in[2];
  const float* Wk = (const float*)d_in[3];
  const float* bk = (const float*)d_in[4];
  const float* Wv = (const float*)d_in[5];
  const float* bv = (const float*)d_in[6];
  const float* Wo = (const float*)d_in[7];
  const float* bo = (const float*)d_in[8];
  float* out = (float*)d_out;

  dim3 grid(2048);   // one block per patch (2 batches x 32 x 32 patches)
  dim3 block(512);   // 16 waves (wave32)
  hipLaunchKernelGGL(patch_attn_kernel, grid, block, 0, stream,
                     x, Wq, bq, Wk, bk, Wv, bv, Wo, bo, out);
}
